// biLSTM_27539330301973
// MI455X (gfx1250) — matrix operand order repaired and checked by reading, not compiled, not run on hardware
//
#include <hip/hip_runtime.h>
#include <type_traits>

typedef __attribute__((ext_vector_type(16))) _Float16 v16h;
typedef __attribute__((ext_vector_type(8)))  float    v8f;

constexpr int B_   = 256;
constexpr int T_   = 4096;
constexpr int H_   = 25;
constexpr int G4_  = 100;   // 4*H gate columns
constexpr int NCOL = 112;   // padded to 7 WMMA N-tiles
constexpr int NT_  = 7;     // number of 16-wide N tiles
constexpr int NTHR = 512;   // 16 waves: waves 0-6 do WMMA, 400 threads own gates

// Fast device math: HW v_rcp_f32 / v_tanh_f32 (both confirmed on gfx1250).
__device__ __forceinline__ float fast_rcp(float x) {
#if __has_builtin(__builtin_amdgcn_rcpf)
    return __builtin_amdgcn_rcpf(x);
#else
    return 1.0f / x;
#endif
}
__device__ __forceinline__ float fast_sigmoid(float x) {
    return fast_rcp(1.0f + __expf(-x));
}
__device__ __forceinline__ float fast_tanh(float x) {
#if __has_builtin(__builtin_amdgcn_tanhf)
    return __builtin_amdgcn_tanhf(x);
#elif __has_builtin(__builtin_amdgcn_tanh_f32)
    return __builtin_amdgcn_tanh_f32(x);
#else
    return 2.0f * fast_rcp(1.0f + __expf(-2.0f * x)) - 1.0f;
#endif
}

// One direction of one LSTM layer for a 16-row batch tile, scanned over time.
//   IN_DIM : input feature count (25 for layer0, 50 for layer1)
//   KCH    : number of K=32 WMMA chunks covering IN_DIM+H (2 for l0, 3 for l1)
//   X_F16  : input sequence is f16 (layer1 reads layer0's f16 output)
//   STORE_SEQ : write full h sequence (layer0) vs only last-step h (layer1)
template <int IN_DIM, int KCH, bool X_F16, bool STORE_SEQ>
__global__ __launch_bounds__(NTHR) void lstm_scan_kernel(
    const void* __restrict__ xin,                // [B,T,IN_DIM] f32 or f16
    const float* __restrict__ w_ih_f, const float* __restrict__ w_hh_f,
    const float* __restrict__ b_ih_f, const float* __restrict__ b_hh_f,
    const float* __restrict__ w_ih_b, const float* __restrict__ w_hh_b,
    const float* __restrict__ b_ih_b, const float* __restrict__ b_hh_b,
    _Float16* __restrict__ hseq_out,             // [B,T,2H] f16 (layer0)
    float* __restrict__ hlast_out,               // [B,2H]   f32 (layer1)
    int nsteps, int dir_override)
{
    constexpr int KPAD = KCH * 32;
    // loader element type: f32 scalar (layer0) or packed f16 pair as b32 (layer1)
    using XT = std::conditional_t<X_F16, unsigned int, float>;

    __shared__ _Float16 wk[KPAD][NCOL];     // combined padded [W_ih; W_hh], K-major
    __shared__ float    bias[NCOL];         // b_ih + b_hh (folded into WMMA acc init)
    __shared__ _Float16 a_buf[16][KPAD];    // A rows: [x_t | h_{t-1} | 0-pad]
    __shared__ float    g_buf[16][NCOL];    // gate pre-activations (bias included)

    const int tid  = threadIdx.x;
    const int wave = tid >> 5;
    const int lane = tid & 31;
    const int b0   = blockIdx.x * 16;
    const int dir  = (dir_override >= 0) ? dir_override : (int)blockIdx.y;

    const float* w_ih = dir ? w_ih_b : w_ih_f;
    const float* w_hh = dir ? w_hh_b : w_hh_f;
    const float* b_ih = dir ? b_ih_b : b_ih_f;
    const float* b_hh = dir ? b_hh_b : b_hh_f;

    // ---- one-time LDS setup: padded combined weight matrix, bias, zeroed A ----
    for (int idx = tid; idx < KPAD * NCOL; idx += NTHR) {
        int k = idx / NCOL, n = idx % NCOL;
        float v = 0.0f;
        if (n < G4_) {
            if (k < IN_DIM)            v = w_ih[n * IN_DIM + k];
            else if (k < IN_DIM + H_)  v = w_hh[n * H_ + (k - IN_DIM)];
        }
        wk[k][n] = (_Float16)v;
    }
    for (int n = tid; n < NCOL; n += NTHR)
        bias[n] = (n < G4_) ? (b_ih[n] + b_hh[n]) : 0.0f;
    for (int idx = tid; idx < 16 * KPAD; idx += NTHR)
        a_buf[idx / KPAD][idx % KPAD] = (_Float16)0.0f;   // zeros h-part & K padding
    __syncthreads();

    // ---- hoist B fragments (weights) + per-column bias into registers ----
    // B k order used here (contiguous, same as A): lanes 0-15 -> N=lane, K=kc*32+0..15;
    //                               lanes 16-31 -> N=lane-16, K=kc*32+16..31.
    v16h  bfrag[KCH];
    float bcol = 0.0f;   // bias for this lane's output column (same for all 8 D rows)
    if (wave < NT_) {
        const int ncol  = wave * 16 + (lane & 15);
        const int kbase = (lane >> 4) * 16;
#pragma unroll
        for (int c = 0; c < KCH; ++c)
#pragma unroll
            for (int j = 0; j < 16; ++j)
                bfrag[c][j] = wk[c * 32 + kbase + j][ncol];
        bcol = bias[ncol];
    }

    // ---- fixed ownership: thread tid -> (eb, eu); owners are also x-loaders ----
    const bool owner = (tid < 16 * H_);                  // 400 owner/loader threads
    const int  eb    = tid / H_;                         // hoisted out of scan loop
    const int  eu    = tid - eb * H_;
    float      creg  = 0.0f;                             // cell state lives in VGPR

    const int t0 = dir ? (T_ - 1) : 0;

    // incremental pointers (no per-step 64-bit address regeneration)
    const XT* xptr = nullptr;        // this thread's x element, advanced each load
    int       xstep;                 // elements of XT per timestep (signed)
    _Float16* hptr = nullptr;        // layer0 hseq output pointer
    if (owner) {
        if constexpr (X_F16) {
            // f16 pair (2*eu, 2*eu+1) of row -> one aligned b32
            const _Float16* xp = (const _Float16*)xin;
            xptr  = (const XT*)(xp + ((size_t)(b0 + eb) * T_ + t0) * IN_DIM + 2 * eu);
            xstep = (dir ? -IN_DIM : IN_DIM) / 2;        // dwords per row
        } else {
            const float* xp = (const float*)xin;
            xptr  = (const XT*)(xp + ((size_t)(b0 + eb) * T_ + t0) * IN_DIM + eu);
            xstep = dir ? -IN_DIM : IN_DIM;
        }
        if constexpr (STORE_SEQ)
            hptr = hseq_out + ((size_t)(b0 + eb) * T_ + t0) * (2 * H_) + dir * H_ + eu;
    }
    const int hstep = dir ? -(2 * H_) : (2 * H_);

    // store one pipelined x value into this thread's A slot
    auto store_x = [&](XT v) {
        if constexpr (X_F16)
            *reinterpret_cast<unsigned int*>(&a_buf[eb][2 * eu]) = v;  // 4B aligned
        else
            a_buf[eb][eu] = (_Float16)v;
    };

    // ---- prologue of the 2-deep load pipeline ----
    XT xval = XT(0);
    if (owner) {
        XT v0 = *xptr; xptr += xstep;        // x(t0)
        store_x(v0);
        if (nsteps > 1) { xval = *xptr; xptr += xstep; }   // issue x(t1)
    }

    // ---- sequential scan: 2 barriers per step ----
    for (int s = 0; s < nsteps; ++s) {
        __syncthreads();             // x_t staged, h_{t-1} written -> A ready

        // G[16,112] = bias + A[16,KPAD] x W[KPAD,112], chained f16 WMMAs (f32 acc)
        if (wave < NT_) {
            v8f acc;
#pragma unroll
            for (int r = 0; r < 8; ++r) acc[r] = bcol;   // bias folded into C
#pragma unroll
            for (int c = 0; c < KCH; ++c) {
                // A k order used here (contiguous, same as B): lane<16 -> row=lane,   K=0-15
                //                               lane>=16 -> row=lane-16, K=16-31
                v16h afrag;
                const int m    = lane & 15;
                const int kadd = (lane >> 4) * 16;
#pragma unroll
                for (int j = 0; j < 16; ++j) {
                    int k = c * 32 + j + kadd;
                    afrag[j] = a_buf[m][k];
                }
                acc = __builtin_amdgcn_wmma_f32_16x16x32_f16(
                    /*neg_a=*/false, afrag, /*neg_b=*/false, bfrag[c],
                    /*c_mod=*/(short)0, acc, /*reuse_a=*/false, /*reuse_b=*/false);
            }
            // D layout: VGPR r -> row r (lanes 0-15) / r+8 (lanes 16-31), col=lane&15
            const int n     = wave * 16 + (lane & 15);
            const int mbase = (lane >> 4) * 8;
#pragma unroll
            for (int r = 0; r < 8; ++r)
                g_buf[mbase + r][n] = acc[r];
        }
        __syncthreads();             // G ready; WMMA done reading a_buf

        if (owner) {
            // drain pipeline: store x(t_{s+1}) (loaded a full step ago), issue x(t_{s+2})
            if (s + 1 < nsteps) {
                store_x(xval);
                if (s + 2 < nsteps) { xval = *xptr; xptr += xstep; }
            }
            // prefetch 4 steps ahead to pre-warm L2 beyond the register pipeline
            if (tid < 16 && s + 4 < nsteps)
                __builtin_prefetch(xptr + 2 * xstep, 0, 0);

            // gate nonlinearity + state update (PyTorch gate order i,f,g,o)
            float gi = g_buf[eb][eu];
            float gf = g_buf[eb][H_ + eu];
            float gg = g_buf[eb][2 * H_ + eu];
            float go = g_buf[eb][3 * H_ + eu];
            creg     = fast_sigmoid(gf) * creg + fast_sigmoid(gi) * fast_tanh(gg);
            float h  = fast_sigmoid(go) * fast_tanh(creg);
            a_buf[eb][IN_DIM + eu] = (_Float16)h;   // feeds next step's A
            if constexpr (STORE_SEQ) {
                *hptr = (_Float16)h;
                hptr += hstep;
            } else {
                if (s == nsteps - 1)
                    hlast_out[(b0 + eb) * (2 * H_) + dir * H_ + eu] = h;
            }
        }
    }
}

// out[b] = sigmoid(relu(relu(hlast[b,:]) . fc_w + fc_b))
__global__ __launch_bounds__(256) void fc_out_kernel(
    const float* __restrict__ hlast, const float* __restrict__ fc_w,
    const float* __restrict__ fc_b, float* __restrict__ out)
{
    int b = blockIdx.x * blockDim.x + threadIdx.x;
    if (b >= B_) return;
    float s = 0.0f;
#pragma unroll
    for (int j = 0; j < 2 * H_; ++j) {
        float h = hlast[b * 2 * H_ + j];
        h = h > 0.0f ? h : 0.0f;
        s += h * fc_w[j];
    }
    s += fc_b[0];
    s = s > 0.0f ? s : 0.0f;
    out[b] = fast_sigmoid(s);
}

extern "C" void kernel_launch(void* const* d_in, const int* in_sizes, int n_in,
                              void* d_out, int out_size, void* d_ws, size_t ws_size,
                              hipStream_t stream)
{
    const float* x        = (const float*)d_in[0];
    const float* w_ih_l0f = (const float*)d_in[1];
    const float* w_hh_l0f = (const float*)d_in[2];
    const float* b_ih_l0f = (const float*)d_in[3];
    const float* b_hh_l0f = (const float*)d_in[4];
    const float* w_ih_l0b = (const float*)d_in[5];
    const float* w_hh_l0b = (const float*)d_in[6];
    const float* b_ih_l0b = (const float*)d_in[7];
    const float* b_hh_l0b = (const float*)d_in[8];
    const float* w_ih_l1f = (const float*)d_in[9];
    const float* w_hh_l1f = (const float*)d_in[10];
    const float* b_ih_l1f = (const float*)d_in[11];
    const float* b_hh_l1f = (const float*)d_in[12];
    const float* w_ih_l1b = (const float*)d_in[13];
    const float* w_hh_l1b = (const float*)d_in[14];
    const float* b_ih_l1b = (const float*)d_in[15];
    const float* b_hh_l1b = (const float*)d_in[16];
    const float* fc2_w    = (const float*)d_in[17];
    const float* fc2_b    = (const float*)d_in[18];

    // workspace layout
    _Float16* hseq  = (_Float16*)d_ws;                               // [B,T,2H] f16
    float*    hlast = (float*)((char*)d_ws +
                       (size_t)B_ * T_ * 2 * H_ * sizeof(_Float16)); // [B,2H] f32

    dim3 blk(NTHR);

    // layer 0: both directions (blockIdx.y), full scan, store f16 sequence
    lstm_scan_kernel<25, 2, false, true><<<dim3(B_ / 16, 2), blk, 0, stream>>>(
        (const void*)x,
        w_ih_l0f, w_hh_l0f, b_ih_l0f, b_hh_l0f,
        w_ih_l0b, w_hh_l0b, b_ih_l0b, b_hh_l0b,
        hseq, nullptr, T_, /*dir_override=*/-1);

    // layer 1 forward: full scan, keep only final h
    lstm_scan_kernel<50, 3, true, false><<<dim3(B_ / 16, 1), blk, 0, stream>>>(
        (const void*)hseq,
        w_ih_l1f, w_hh_l1f, b_ih_l1f, b_hh_l1f,
        w_ih_l1f, w_hh_l1f, b_ih_l1f, b_hh_l1f,   // unused (dir=0)
        nullptr, hlast, T_, /*dir_override=*/0);

    // layer 1 backward: output at t=T-1 is the reverse scan's FIRST step -> 1 step
    lstm_scan_kernel<50, 3, true, false><<<dim3(B_ / 16, 1), blk, 0, stream>>>(
        (const void*)hseq,
        w_ih_l1b, w_hh_l1b, b_ih_l1b, b_hh_l1b,   // unused (dir=1)
        w_ih_l1b, w_hh_l1b, b_ih_l1b, b_hh_l1b,
        nullptr, hlast, /*nsteps=*/1, /*dir_override=*/1);

    // final FC + relu + sigmoid
    fc_out_kernel<<<dim3(1), dim3(256), 0, stream>>>(hlast, fc2_w, fc2_b, (float*)d_out);
}
